// trajectory2seq_35553739276789
// MI455X (gfx1250) — matrix operand order repaired and checked by reading, not compiled, not run on hardware
//
#include <hip/hip_runtime.h>
#include <math.h>

// Problem constants (match reference)
#define B_     512
#define L_     128
#define F_     512
#define H_     512
#define V_     128
#define TDEC_  64
#define KC_    128   // K-chunk staged per async pipeline stage in the GRU step

typedef __attribute__((ext_vector_type(16))) __bf16 v16bf;
typedef __attribute__((ext_vector_type(8)))  __bf16 v8bf;
typedef __attribute__((ext_vector_type(8)))  float  v8f;

// ---------------------------------------------------------------------------
// WMMA fragment loaders (wave32, 16x16x32 bf16).
// A (16x32, MxK), ISA 7.12.2: lane l holds row m=l&15;
//   elems 0..7  -> K = (l>>4)*8 + i ; elems 8..15 -> K = 16 + (l>>4)*8 + (i-8)
// B (32x16, KxN): lane l holds col n=l&15; same K order as A:
//   elems 0..7  -> K = (l>>4)*8 + i ; elems 8..15 -> K = 16 + (l>>4)*8 + (i-8)
// All GEMMs are X @ W^T with row-major W, so B-column n == W row n.
// Loaders work on global or LDS pointers (provenance selects global_load_b128
// vs ds_load_b128).
// ---------------------------------------------------------------------------
__device__ __forceinline__ v16bf load_a_bf16(const __bf16* __restrict__ row, int k0, int kh) {
  v8bf a0 = *(const v8bf*)(row + k0 + kh * 8);
  v8bf a1 = *(const v8bf*)(row + k0 + 16 + kh * 8);
  v16bf r;
#pragma unroll
  for (int i = 0; i < 8; ++i) { r[i] = a0[i]; r[i + 8] = a1[i]; }
  return r;
}

__device__ __forceinline__ v16bf load_a_f32(const float* __restrict__ row, int k0, int kh) {
  v8f a0 = *(const v8f*)(row + k0 + kh * 8);
  v8f a1 = *(const v8f*)(row + k0 + 16 + kh * 8);
  v16bf r;
#pragma unroll
  for (int i = 0; i < 8; ++i) { r[i] = (__bf16)a0[i]; r[i + 8] = (__bf16)a1[i]; }
  return r;
}

__device__ __forceinline__ v16bf load_b_bf16(const __bf16* __restrict__ wrow, int k0, int kh) {
  v8bf a0 = *(const v8bf*)(wrow + k0 + kh * 8);
  v8bf a1 = *(const v8bf*)(wrow + k0 + 16 + kh * 8);
  v16bf r;
#pragma unroll
  for (int i = 0; i < 8; ++i) { r[i] = a0[i]; r[i + 8] = a1[i]; }
  return r;
}

__device__ __forceinline__ v8f wmma_bf16(v16bf a, v16bf b, v8f c) {
  return __builtin_amdgcn_wmma_f32_16x16x32_bf16(false, a, false, b, (short)0, c, false, false);
}

__device__ __forceinline__ float sigmoidf_(float x) { return 1.0f / (1.0f + __expf(-x)); }

// --- CDNA5 async global->LDS copy (ASYNCcnt), CDNA5 ISA ---
// 16 bytes per lane: LDS byte offset in a VGPR, 64-bit global address in a pair.
__device__ __forceinline__ void async_copy16(unsigned lds_off, const void* gsrc) {
  asm volatile("global_load_async_to_lds_b128 %0, %1, off"
               :: "v"(lds_off), "v"(gsrc) : "memory");
}
__device__ __forceinline__ void wait_async0() {
  asm volatile("s_wait_asynccnt 0" ::: "memory");
}
__device__ __forceinline__ unsigned lds_off_of(const void* p) {
  return (unsigned)(size_t)p;  // generic LDS pointer: offset in addr[31:0]
}

// ---------------------------------------------------------------------------
// Utility kernels
// ---------------------------------------------------------------------------
__global__ void cvt_f32_to_bf16(const float* __restrict__ s, __bf16* __restrict__ d, int n) {
  int i = blockIdx.x * blockDim.x + threadIdx.x;
  if (i < n) d[i] = (__bf16)s[i];
}

// x (B,L,F) f32 -> (L,B,F) bf16: per-step working set becomes one contiguous
// 512 KB block in L2, and the per-use v_cvt chains disappear from the scan.
__global__ void cvt_x_tmajor(const float* __restrict__ s, __bf16* __restrict__ d) {
  const int i = blockIdx.x * blockDim.x + threadIdx.x;  // < B_*L_*F_ = 2^25
  const int f = i & (F_ - 1);
  const int t = (i >> 9) & (L_ - 1);   // F_ = 2^9
  const int b = i >> 16;               // L_*F_ = 2^16
  d[((size_t)t * B_ + b) * F_ + f] = (__bf16)s[i];
}

__global__ void fill_zero_u32(unsigned* __restrict__ p, int n) {
  int i = blockIdx.x * blockDim.x + threadIdx.x;
  if (i < n) p[i] = 0u;
}

// ---------------------------------------------------------------------------
// One bidirectional GRU time step, fully fused, with an async-LDS weight
// pipeline: the 8 waves of a block share one 16-column strip, so the 6 gate
// weight tiles (3 gates x {Wih,Whh}) are staged ONCE per block into LDS,
// double-buffered in K-chunks of 128 and prefetched with
// global_load_async_to_lds_b128 while WMMAs consume the current chunk.
// XBF16: A-operand x is pre-converted bf16 (L,B,F); else f32 (B,L,F) with
// in-register conversion (small-workspace fallback).
// grid = (128, 2): blockIdx.x -> {col strip (32), row group (4)}, y = dir.
// ---------------------------------------------------------------------------
template <bool XBF16>
__global__ void __launch_bounds__(256) gru_step_kernel(
    const float*  __restrict__ x,      // (B,L,F) f32        (XBF16=false)
    const __bf16* __restrict__ xb,     // (L,B,F) bf16       (XBF16=true)
    const __bf16* __restrict__ Wih,    // (2,3H,F) bf16
    const __bf16* __restrict__ Whh,    // (2,3H,H) bf16
    const float*  __restrict__ bih,    // (2,3H)
    const float*  __restrict__ bhh,    // (2,3H)
    const float*  __restrict__ h_in_f, // (2,B,H)
    const __bf16* __restrict__ h_in_b, // (2,B,H)
    float*        __restrict__ h_out_f,
    __bf16*       __restrict__ h_out_b,
    int t) {
  __shared__ __bf16 wlds[2][6][16][KC_];  // 48 KB: [buf][mat][ncol][k]

  const int lane = threadIdx.x & 31;
  const int wave = threadIdx.x >> 5;
  const int dir  = blockIdx.y;
  const int tn   = (blockIdx.x >> 2) << 4;                 // column strip
  const int tm   = ((blockIdx.x & 3) * 8 + wave) << 4;     // row tile
  const int kh   = lane >> 4;
  const int lm   = lane & 15;

  const int tcur = (dir == 0) ? t : (L_ - 1 - t);

  const float*  xrow  = x + ((size_t)(tm + lm) * L_ + tcur) * F_;
  const __bf16* xbrow = xb + ((size_t)tcur * B_ + tm + lm) * F_;
  const __bf16* hrow  = h_in_b + ((size_t)dir * B_ + tm + lm) * H_;
  const size_t  g3    = (size_t)dir * 3 * H_;

  const __bf16* wbase[6];
#pragma unroll
  for (int g = 0; g < 3; ++g) {
    wbase[g]     = Wih + (g3 + (size_t)g * H_ + tn) * F_;
    wbase[3 + g] = Whh + (g3 + (size_t)g * H_ + tn) * H_;
  }

  // Stage one K-chunk of all 6 weight tiles: 256 threads x 6 x 16B.
  const int srow = threadIdx.x >> 4;  // 0..15 (weight row == output col)
  const int svec = threadIdx.x & 15;  // 0..15 (16B vector within chunk)
  auto stage = [&](int buf, int c) {
#pragma unroll
    for (int m = 0; m < 6; ++m) {
      const __bf16* src = wbase[m] + (size_t)srow * 512 + c * KC_ + svec * 8;
      async_copy16(lds_off_of(&wlds[buf][m][srow][svec * 8]), src);
    }
  };

  stage(0, 0);
  wait_async0();
  __syncthreads();

  v8f accR = {}, accZ = {}, accXN = {}, accHN = {};
#pragma unroll
  for (int c = 0; c < F_ / KC_; ++c) {
    const int cur = c & 1;
    if (c + 1 < F_ / KC_) stage(1 - cur, c + 1);  // prefetch next chunk
#pragma unroll 2
    for (int kl = 0; kl < KC_; kl += 32) {
      const int kg = c * KC_ + kl;
      v16bf ax;
      if constexpr (XBF16) ax = load_a_bf16(xbrow, kg, kh);
      else                 ax = load_a_f32(xrow, kg, kh);
      v16bf ah = load_a_bf16(hrow, kg, kh);
      accR  = wmma_bf16(ax, load_b_bf16(&wlds[cur][0][lm][0], kl, kh), accR);
      accR  = wmma_bf16(ah, load_b_bf16(&wlds[cur][3][lm][0], kl, kh), accR);
      accZ  = wmma_bf16(ax, load_b_bf16(&wlds[cur][1][lm][0], kl, kh), accZ);
      accZ  = wmma_bf16(ah, load_b_bf16(&wlds[cur][4][lm][0], kl, kh), accZ);
      accXN = wmma_bf16(ax, load_b_bf16(&wlds[cur][2][lm][0], kl, kh), accXN);
      accHN = wmma_bf16(ah, load_b_bf16(&wlds[cur][5][lm][0], kl, kh), accHN);
    }
    wait_async0();
    __syncthreads();
  }

  const int   ncol = tn + lm;
  const float bR   = bih[g3 + ncol] + bhh[g3 + ncol];
  const float bZ   = bih[g3 + H_ + ncol] + bhh[g3 + H_ + ncol];
  const float bXN  = bih[g3 + 2 * H_ + ncol];
  const float bHN  = bhh[g3 + 2 * H_ + ncol];

#pragma unroll
  for (int i = 0; i < 8; ++i) {
    const int    mrow = tm + kh * 8 + i;  // C/D: VGPR i -> M = (lane>=16?8:0)+i
    const size_t idx  = (size_t)dir * B_ * H_ + (size_t)mrow * H_ + ncol;
    const float  hp   = h_in_f[idx];
    const float  r    = sigmoidf_(accR[i] + bR);
    const float  z    = sigmoidf_(accZ[i] + bZ);
    const float  n    = tanhf(accXN[i] + bXN + r * (accHN[i] + bHN));
    const float  hn   = (1.0f - z) * n + z * hp;
    h_out_f[idx] = hn;
    h_out_b[idx] = (__bf16)hn;
  }
}

// ---------------------------------------------------------------------------
// hidden = relu( view(h_enc,(B,2H)) @ enc2dec_W^T + b )
// hcat read flat as (B,2H) reproduces torch's batch-mixing .view().
// Weight strip (16 rows x 1024) async-staged into LDS once per block.
// ---------------------------------------------------------------------------
__global__ void __launch_bounds__(256) enc2dec_kernel(
    const __bf16* __restrict__ hcat,  // (2*B*H) flat -> (B, 2H)
    const __bf16* __restrict__ Wb,    // (H, 2H) bf16
    const float*  __restrict__ bias,  // (H)
    float*  __restrict__ h_out_f,     // (B,H)
    __bf16* __restrict__ h_out_b) {
  __shared__ __bf16 wlds[16][2 * H_];  // 32 KB

  const int lane = threadIdx.x & 31;
  const int wave = threadIdx.x >> 5;
  const int tn   = (blockIdx.x >> 2) << 4;
  const int tm   = ((blockIdx.x & 3) * 8 + wave) << 4;
  const int kh   = lane >> 4;
  const int lm   = lane & 15;

#pragma unroll
  for (int j = 0; j < 8; ++j) {  // 16 rows x 128 vec16B = 2048 vecs / 256 thr
    const int v = threadIdx.x + 256 * j;
    const int row = v >> 7, vec = v & 127;
    async_copy16(lds_off_of(&wlds[row][vec * 8]),
                 Wb + (size_t)(tn + row) * (2 * H_) + vec * 8);
  }
  wait_async0();
  __syncthreads();

  const __bf16* arow = hcat + (size_t)(tm + lm) * (2 * H_);

  v8f acc = {};
#pragma unroll 4
  for (int k0 = 0; k0 < 2 * H_; k0 += 32)
    acc = wmma_bf16(load_a_bf16(arow, k0, kh), load_b_bf16(&wlds[lm][0], k0, kh), acc);

  const int   ncol = tn + lm;
  const float bv   = bias[ncol];
#pragma unroll
  for (int i = 0; i < 8; ++i) {
    const int    mrow = tm + kh * 8 + i;
    const size_t idx  = (size_t)mrow * H_ + ncol;
    float v = acc[i] + bv;
    v = v > 0.0f ? v : 0.0f;
    h_out_f[idx] = v;
    h_out_b[idx] = (__bf16)v;
  }
}

// ---------------------------------------------------------------------------
// Decoder RNN step: h' = tanh(emb[tok] @ Wih^T + h @ Whh^T + bih + bhh)
// Both 16-row weight strips async-staged into LDS once per block.
// ---------------------------------------------------------------------------
__global__ void __launch_bounds__(256) dec_rnn_kernel(
    const int*    __restrict__ tok,   // (B)
    const __bf16* __restrict__ embb,  // (V,H)
    const __bf16* __restrict__ Wihb,  // (H,H)
    const __bf16* __restrict__ Whhb,  // (H,H)
    const float*  __restrict__ bih,
    const float*  __restrict__ bhh,
    const __bf16* __restrict__ h_in_b,  // (B,H)
    float*  __restrict__ h_out_f,
    __bf16* __restrict__ h_out_b) {
  __shared__ __bf16 wlds[2][16][H_];  // 32 KB

  const int lane = threadIdx.x & 31;
  const int wave = threadIdx.x >> 5;
  const int tn   = (blockIdx.x >> 2) << 4;
  const int tm   = ((blockIdx.x & 3) * 8 + wave) << 4;
  const int kh   = lane >> 4;
  const int lm   = lane & 15;

#pragma unroll
  for (int m = 0; m < 2; ++m) {
    const __bf16* W = m ? Whhb : Wihb;
#pragma unroll
    for (int j = 0; j < 4; ++j) {  // 16 rows x 64 vec16B = 1024 vecs / 256 thr
      const int v = threadIdx.x + 256 * j;
      const int row = v >> 6, vec = v & 63;
      async_copy16(lds_off_of(&wlds[m][row][vec * 8]),
                   W + (size_t)(tn + row) * H_ + vec * 8);
    }
  }
  wait_async0();
  __syncthreads();

  const __bf16* erow = embb + (size_t)tok[tm + lm] * H_;  // embedding gather
  const __bf16* hrow = h_in_b + (size_t)(tm + lm) * H_;

  v8f acc = {};
#pragma unroll 4
  for (int k0 = 0; k0 < H_; k0 += 32) {
    acc = wmma_bf16(load_a_bf16(erow, k0, kh), load_b_bf16(&wlds[0][lm][0], k0, kh), acc);
    acc = wmma_bf16(load_a_bf16(hrow, k0, kh), load_b_bf16(&wlds[1][lm][0], k0, kh), acc);
  }

  const int   ncol = tn + lm;
  const float bv   = bih[ncol] + bhh[ncol];
#pragma unroll
  for (int i = 0; i < 8; ++i) {
    const int    mrow = tm + kh * 8 + i;
    const size_t idx  = (size_t)mrow * H_ + ncol;
    const float  v    = tanhf(acc[i] + bv);
    h_out_f[idx] = v;
    h_out_b[idx] = (__bf16)v;
  }
}

// ---------------------------------------------------------------------------
// logits = h' @ fc_W^T + fc_b -> d_out[:, t, :]; argmax(first-max) -> tok'
// The 16-row A (hidden) tile is shared by all 8 waves -> async-staged in LDS;
// logits staged in LDS for the row-wise argmax.
// ---------------------------------------------------------------------------
__global__ void __launch_bounds__(256) dec_logits_kernel(
    const __bf16* __restrict__ h_b,   // (B,H)
    const __bf16* __restrict__ fcWb,  // (V,H)
    const float*  __restrict__ fcb,   // (V)
    float* __restrict__ out,          // (B,TDEC,V) region of d_out
    int*   __restrict__ tok_out,      // (B)
    int t) {
  __shared__ __bf16 atile[16][H_];   // 16 KB
  __shared__ float  lrow[16 * V_];   //  8 KB

  const int lane = threadIdx.x & 31;
  const int wave = threadIdx.x >> 5;
  const int tm   = blockIdx.x * 16;
  const int tn   = wave * 16;  // 8 waves * 16 = V_
  const int kh   = lane >> 4;
  const int lm   = lane & 15;

#pragma unroll
  for (int j = 0; j < 4; ++j) {  // 16 rows x 64 vec16B = 1024 vecs / 256 thr
    const int v = threadIdx.x + 256 * j;
    const int row = v >> 6, vec = v & 63;
    async_copy16(lds_off_of(&atile[row][vec * 8]),
                 h_b + (size_t)(tm + row) * H_ + vec * 8);
  }
  wait_async0();
  __syncthreads();

  const __bf16* brow = fcWb + (size_t)(tn + lm) * H_;

  v8f acc = {};
#pragma unroll 4
  for (int k0 = 0; k0 < H_; k0 += 32)
    acc = wmma_bf16(load_a_bf16(&atile[lm][0], k0, kh), load_b_bf16(brow, k0, kh), acc);

  const int   ncol = tn + lm;
  const float bv   = fcb[ncol];
#pragma unroll
  for (int i = 0; i < 8; ++i) {
    const int   lr = kh * 8 + i;  // local row 0..15
    const float v  = acc[i] + bv;
    out[(size_t)(tm + lr) * TDEC_ * V_ + (size_t)t * V_ + ncol] = v;
    lrow[lr * V_ + ncol] = v;
  }
  __syncthreads();

  if (threadIdx.x < 16) {
    const float* r  = lrow + threadIdx.x * V_;
    int   best = 0;
    float bvx  = r[0];
    for (int j = 1; j < V_; ++j) {
      const float x = r[j];
      if (x > bvx) { bvx = x; best = j; }  // strict > == first occurrence
    }
    tok_out[tm + threadIdx.x] = best;
  }
}

// ---------------------------------------------------------------------------
// Host orchestration
// ---------------------------------------------------------------------------
extern "C" void kernel_launch(void* const* d_in, const int* in_sizes, int n_in,
                              void* d_out, int out_size, void* d_ws, size_t ws_size,
                              hipStream_t stream) {
  (void)in_sizes; (void)n_in; (void)out_size;

  const float* x    = (const float*)d_in[0];
  const float* gWih = (const float*)d_in[1];
  const float* gWhh = (const float*)d_in[2];
  const float* gbih = (const float*)d_in[3];
  const float* gbhh = (const float*)d_in[4];
  const float* e2dW = (const float*)d_in[5];
  const float* e2db = (const float*)d_in[6];
  const float* emb  = (const float*)d_in[7];
  const float* rWih = (const float*)d_in[8];
  const float* rWhh = (const float*)d_in[9];
  const float* rbih = (const float*)d_in[10];
  const float* rbhh = (const float*)d_in[11];
  const float* fcW  = (const float*)d_in[12];
  const float* fcb  = (const float*)d_in[13];
  // d_in[14] = maxlen_trad (== TDEC_, compile-time)

  // ---- workspace layout (~18 MB base + optional 64 MB bf16 x) ----
  char* p = (char*)d_ws;
  auto alloc = [&](size_t bytes) -> char* {
    char* r = p;
    p += (bytes + 255) & ~(size_t)255;
    return r;
  };
  __bf16* gWih_b = (__bf16*)alloc((size_t)2 * 3 * H_ * F_ * 2);
  __bf16* gWhh_b = (__bf16*)alloc((size_t)2 * 3 * H_ * H_ * 2);
  __bf16* e2dW_b = (__bf16*)alloc((size_t)H_ * 2 * H_ * 2);
  __bf16* emb_b  = (__bf16*)alloc((size_t)V_ * H_ * 2);
  __bf16* rWih_b = (__bf16*)alloc((size_t)H_ * H_ * 2);
  __bf16* rWhh_b = (__bf16*)alloc((size_t)H_ * H_ * 2);
  __bf16* fcW_b  = (__bf16*)alloc((size_t)V_ * H_ * 2);
  float*  henc_f[2]; __bf16* henc_b[2];
  float*  hdec_f[2]; __bf16* hdec_b[2];
  int*    tokb[2];
  for (int i = 0; i < 2; ++i) henc_f[i] = (float*)alloc((size_t)2 * B_ * H_ * 4);
  for (int i = 0; i < 2; ++i) henc_b[i] = (__bf16*)alloc((size_t)2 * B_ * H_ * 2);
  for (int i = 0; i < 2; ++i) hdec_f[i] = (float*)alloc((size_t)B_ * H_ * 4);
  for (int i = 0; i < 2; ++i) hdec_b[i] = (__bf16*)alloc((size_t)B_ * H_ * 2);
  for (int i = 0; i < 2; ++i) tokb[i]   = (int*)alloc((size_t)B_ * 4);

  // Optional time-major bf16 copy of x (picked deterministically from ws_size).
  const size_t xbytes = (size_t)B_ * L_ * F_ * 2;
  const size_t used   = (size_t)(p - (char*)d_ws);
  const bool   use_xb = (used + xbytes + 256) <= ws_size;
  __bf16* x_b = use_xb ? (__bf16*)alloc(xbytes) : nullptr;

  // ---- one-time f32 -> bf16 weight conversion (L2-resident thereafter) ----
  auto cvt = [&](const float* s, __bf16* d, int n) {
    cvt_f32_to_bf16<<<(n + 255) / 256, 256, 0, stream>>>(s, d, n);
  };
  cvt(gWih, gWih_b, 2 * 3 * H_ * F_);
  cvt(gWhh, gWhh_b, 2 * 3 * H_ * H_);
  cvt(e2dW, e2dW_b, H_ * 2 * H_);
  cvt(emb,  emb_b,  V_ * H_);
  cvt(rWih, rWih_b, H_ * H_);
  cvt(rWhh, rWhh_b, H_ * H_);
  cvt(fcW,  fcW_b,  V_ * H_);
  if (use_xb)
    cvt_x_tmajor<<<(B_ * L_ * F_) / 256, 256, 0, stream>>>(x, x_b);

  // ---- zero initial state (h0 = 0, tok0 = 0) ----
  auto zero = [&](void* ptr, size_t bytes) {
    int nw = (int)(bytes / 4);
    fill_zero_u32<<<(nw + 255) / 256, 256, 0, stream>>>((unsigned*)ptr, nw);
  };
  zero(henc_f[0], (size_t)2 * B_ * H_ * 4);
  zero(henc_b[0], (size_t)2 * B_ * H_ * 2);
  zero(tokb[0],   (size_t)B_ * 4);

  // ---- bidirectional GRU encoder: 128 sequential fused steps ----
  for (int t = 0; t < L_; ++t) {
    const int a = t & 1, b = 1 - a;
    if (use_xb)
      gru_step_kernel<true><<<dim3(128, 2), 256, 0, stream>>>(
          x, x_b, gWih_b, gWhh_b, gbih, gbhh,
          henc_f[a], henc_b[a], henc_f[b], henc_b[b], t);
    else
      gru_step_kernel<false><<<dim3(128, 2), 256, 0, stream>>>(
          x, x_b, gWih_b, gWhh_b, gbih, gbhh,
          henc_f[a], henc_b[a], henc_f[b], henc_b[b], t);
  }
  // L_ even -> final hidden (both dirs, contiguous (2,B,H)) lands in buffer 0

  // ---- enc2dec projection + ReLU (torch-view batch mixing via flat read) ----
  enc2dec_kernel<<<128, 256, 0, stream>>>(henc_b[0], e2dW_b, e2db,
                                          hdec_f[0], hdec_b[0]);

  // ---- greedy decoder: 64 steps of (RNN update; logits + argmax) ----
  float* out_logits = (float*)d_out;  // (B, TDEC, V)
  for (int t = 0; t < TDEC_; ++t) {
    const int a = t & 1, b = 1 - a;
    dec_rnn_kernel<<<128, 256, 0, stream>>>(
        tokb[a], emb_b, rWih_b, rWhh_b, rbih, rbhh,
        hdec_b[a], hdec_f[b], hdec_b[b]);
    dec_logits_kernel<<<32, 256, 0, stream>>>(
        hdec_b[b], fcW_b, fcb, out_logits, tokb[b], t);
  }

  // ---- h_last (TDEC_ even -> buffer 0) appended after logits in d_out ----
  hipMemcpyAsync((float*)d_out + (size_t)B_ * TDEC_ * V_, hdec_f[0],
                 (size_t)B_ * H_ * sizeof(float), hipMemcpyDeviceToDevice, stream);
}
